// SimpleMoE_60619168416469
// MI455X (gfx1250) — matrix operand order repaired and checked by reading, not compiled, not run on hardware
//
#include <hip/hip_runtime.h>
#include <hip/hip_bf16.h>

#define N_NODES 50000
#define N_EDGES 800000
#define F_IN    128
#define H_DIM   64
#define N_EXP   3
#define N_GRAPH 64
#define N_CLS   3
#define BN_EPS  1e-5f

typedef __attribute__((ext_vector_type(16))) __bf16 v16bf;
typedef __attribute__((ext_vector_type(8)))  float  v8f;

__device__ __forceinline__ unsigned short f2bf(float f) {
    unsigned int u = __float_as_uint(f);
    unsigned int r = u + 0x7FFFu + ((u >> 16) & 1u);   // round-to-nearest-even
    return (unsigned short)(r >> 16);
}

// ---------------- utility kernels ----------------
__global__ void k_zero(float* __restrict__ p, long n) {
    long i = (long)blockIdx.x * blockDim.x + threadIdx.x;
    if (i < n) p[i] = 0.0f;
}

__global__ void k_cvt_bf16(const float* __restrict__ in, unsigned short* __restrict__ out, long n) {
    long i = (long)blockIdx.x * blockDim.x + threadIdx.x;
    if (i < n) out[i] = f2bf(in[i]);
}

// W_exp [NE][F][H] f32 -> Wt [NE][H][F] bf16 (B stored column-major => rows contiguous in K)
__global__ void k_prep_wexp(const float* __restrict__ W, unsigned short* __restrict__ Wt) {
    int t = blockIdx.x * blockDim.x + threadIdx.x;
    if (t >= N_EXP * F_IN * H_DIM) return;
    int e = t / (F_IN * H_DIM);
    int rem = t - e * (F_IN * H_DIM);
    int f = rem / H_DIM, h = rem % H_DIM;
    Wt[(size_t)e * H_DIM * F_IN + (size_t)h * F_IN + f] = f2bf(W[t]);
}

__global__ void k_prep_gw1(const float* __restrict__ W, unsigned short* __restrict__ Wt) {
    int t = blockIdx.x * blockDim.x + threadIdx.x;
    if (t >= F_IN * 32) return;
    int f = t / 32, h = t % 32;
    Wt[(size_t)h * F_IN + f] = f2bf(W[t]);
}

__global__ void k_prep_w2(const float* __restrict__ W, unsigned short* __restrict__ Wt) {
    int t = blockIdx.x * blockDim.x + threadIdx.x;
    if (t >= H_DIM * H_DIM) return;
    int f = t / H_DIM, h = t % H_DIM;
    Wt[(size_t)h * H_DIM + f] = f2bf(W[t]);
}

// ---------------- degree / norm ----------------
__global__ void k_edge_deg(const int* __restrict__ dst, float* __restrict__ deg) {
    int e = blockIdx.x * blockDim.x + threadIdx.x;
    if (e < N_EDGES) atomicAdd(&deg[dst[e]], 1.0f);
}

__global__ void k_deg_final(const float* __restrict__ deg, float* __restrict__ dis, float* __restrict__ dinv) {
    int n = blockIdx.x * blockDim.x + threadIdx.x;
    if (n >= N_NODES) return;
    float d = deg[n] + 1.0f;
    dis[n]  = rsqrtf(d);
    dinv[n] = 1.0f / d;
}

__global__ void k_enorm(const int* __restrict__ src, const int* __restrict__ dst,
                        const float* __restrict__ dis, float* __restrict__ en) {
    int e = blockIdx.x * blockDim.x + threadIdx.x;
    if (e < N_EDGES) en[e] = dis[src[e]] * dis[dst[e]];
}

// ---------------- WMMA bf16 GEMM: C[M,NT*16] = A_bf16[M,K] * Bt_bf16[NT*16,K]^T ----------------
// A row-major bf16 (lda), Bt row-major bf16 (ldb = K stride): Bt[col][k] == B[k][col].
// One wave computes a 16-row block x NT 16-col tiles; KT tiles of K=32.
template <int NT, int KT>
__global__ void k_wmma_gemm(const unsigned short* __restrict__ A, int lda,
                            const unsigned short* __restrict__ Bt, int ldb,
                            float* __restrict__ C, int ldc, int Mtiles,
                            const float* __restrict__ bias, int relu) {
    int wid = blockIdx.x * (blockDim.x >> 5) + (threadIdx.x >> 5);
    if (wid >= Mtiles) return;                 // wave-uniform exit; EXEC stays all-ones
    const int lane = threadIdx.x & 31;
    const int mrow = lane & 15;
    const int kh   = lane >> 4;                // K-half select
    const int row0 = wid << 4;

    v8f acc[NT];
#pragma unroll
    for (int j = 0; j < NT; ++j)
#pragma unroll
        for (int r = 0; r < 8; ++r) acc[j][r] = 0.0f;

#pragma unroll
    for (int kt = 0; kt < KT; ++kt) {
        // A fragment: 16x32 bf16.  VGPR v holds K pair {kb,kb+1}, kb = (v&4?16:0)+(v&3)*2+kh*8
        union { unsigned int u[8]; v16bf v; } fa;
        const unsigned short* ap = A + (size_t)(row0 + mrow) * lda + kt * 32;
#pragma unroll
        for (int r = 0; r < 8; ++r) {
            int kb = ((r & 4) ? 16 : 0) + (r & 3) * 2 + kh * 8;
            fa.u[r] = *(const unsigned int*)(ap + kb);
        }
#pragma unroll
        for (int j = 0; j < NT; ++j) {
            // B fragment: 32x16 bf16; lane = col; VGPR v holds K pair {kb,kb+1}, kb = (v&4?16:0)+(v&3)*2+kh*8
            union { unsigned int u[8]; v16bf v; } fb;
            const unsigned short* bp = Bt + (size_t)(j * 16 + mrow) * ldb + kt * 32;
#pragma unroll
            for (int r = 0; r < 8; ++r) fb.u[r] = *(const unsigned int*)(bp + ((r & 4) ? 16 : 0) + (r & 3) * 2 + kh * 8);
            acc[j] = __builtin_amdgcn_wmma_f32_16x16x32_bf16(
                false, fa.v, false, fb.v, (short)0, acc[j], false, false);
        }
    }

    const int rb = kh * 8;                     // C layout: VGPR r -> row rb+r, col = lane&15
#pragma unroll
    for (int j = 0; j < NT; ++j) {
        float bcol = bias ? bias[j * 16 + mrow] : 0.0f;
#pragma unroll
        for (int r = 0; r < 8; ++r) {
            float val = acc[j][r] + bcol;
            if (relu) val = fmaxf(val, 0.0f);
            C[(size_t)(row0 + rb + r) * ldc + j * 16 + mrow] = val;
        }
    }
}

// ---------------- gate layer 2 + softmax ----------------
__global__ void k_gate2(const float* __restrict__ g1, const float* __restrict__ W2g,
                        const float* __restrict__ b2g, float* __restrict__ gate) {
    int n = blockIdx.x * blockDim.x + threadIdx.x;
    if (n >= N_NODES) return;
    const float* r = g1 + (size_t)n * 32;
    float l0 = b2g[0], l1 = b2g[1], l2 = b2g[2];
#pragma unroll
    for (int k = 0; k < 32; ++k) {
        float v = r[k];
        l0 += v * W2g[k * 3 + 0];
        l1 += v * W2g[k * 3 + 1];
        l2 += v * W2g[k * 3 + 2];
    }
    float m = fmaxf(l0, fmaxf(l1, l2));
    float e0 = __expf(l0 - m), e1 = __expf(l1 - m), e2 = __expf(l2 - m);
    float inv = 1.0f / (e0 + e1 + e2);
    gate[n * 3 + 0] = e0 * inv;
    gate[n * 3 + 1] = e1 * inv;
    gate[n * 3 + 2] = e2 * inv;
}

// ---------------- edge scatter (GCN aggregation), C channels (mult of 4) ----------------
__global__ void k_scatter(const int* __restrict__ src, const int* __restrict__ dst,
                          const float* __restrict__ enorm, const float* __restrict__ hin,
                          float* __restrict__ agg, int C) {
    long tid = (long)blockIdx.x * blockDim.x + threadIdx.x;
    int C4 = C >> 2;
    long total = (long)N_EDGES * C4;
    if (tid >= total) return;
    int e  = (int)(tid / C4);
    int c0 = (int)(tid % C4) * 4;
    int s = src[e], d = dst[e];
    float nrm = enorm[e];
    const float4 v = *(const float4*)(hin + (size_t)s * C + c0);
    float* a = agg + (size_t)d * C + c0;
    atomicAdd(a + 0, v.x * nrm);
    atomicAdd(a + 1, v.y * nrm);
    atomicAdd(a + 2, v.z * nrm);
    atomicAdd(a + 3, v.w * nrm);
}

// ---------------- MoE gated combine (+ self loop + bias) ----------------
__global__ void k_combine1(const float* __restrict__ agg, const float* __restrict__ h1,
                           const float* __restrict__ gate, const float* __restrict__ dinv,
                           const float* __restrict__ bexp, float* __restrict__ hc) {
    int i = blockIdx.x * blockDim.x + threadIdx.x;
    if (i >= N_NODES * H_DIM) return;
    int n = i >> 6, h = i & 63;
    float di = dinv[n];
    float s = 0.0f;
#pragma unroll
    for (int e = 0; e < N_EXP; ++e) {
        size_t idx = (size_t)n * 192 + e * 64 + h;
        float v = agg[idx] + h1[idx] * di + bexp[e * 64 + h];
        s += gate[n * 3 + e] * v;
    }
    hc[i] = s;
}

__global__ void k_combine2(const float* __restrict__ agg2, const float* __restrict__ h3,
                           const float* __restrict__ dinv, const float* __restrict__ b2,
                           float* __restrict__ h4) {
    int i = blockIdx.x * blockDim.x + threadIdx.x;
    if (i >= N_NODES * H_DIM) return;
    int n = i >> 6, h = i & 63;
    h4[i] = agg2[i] + h3[i] * dinv[n] + b2[h];
}

// ---------------- batch norm stats (launch with total threads % 64 == 0) ----------------
__global__ void k_bnstats(const float* __restrict__ in, float* __restrict__ s_sum,
                          float* __restrict__ s_sq) {
    int tid = blockIdx.x * blockDim.x + threadIdx.x;
    int T = gridDim.x * blockDim.x;
    float s = 0.0f, q = 0.0f;
    for (long i = tid; i < (long)N_NODES * H_DIM; i += T) {
        float v = in[i];
        s += v; q += v * v;
    }
    int c = tid & (H_DIM - 1);
    atomicAdd(&s_sum[c], s);
    atomicAdd(&s_sq[c], q);
}

__global__ void k_bnapply_bf16(const float* __restrict__ in, const float* __restrict__ s_sum,
                               const float* __restrict__ s_sq, const float* __restrict__ g,
                               const float* __restrict__ b, unsigned short* __restrict__ out) {
    int i = blockIdx.x * blockDim.x + threadIdx.x;
    if (i >= N_NODES * H_DIM) return;
    int c = i & 63;
    float mu  = s_sum[c] * (1.0f / N_NODES);
    float var = s_sq[c] * (1.0f / N_NODES) - mu * mu;
    float y = (in[i] - mu) * rsqrtf(var + BN_EPS) * g[c] + b[c];
    out[i] = f2bf(fmaxf(y, 0.0f));
}

__global__ void k_bnapply_f32(const float* __restrict__ in, const float* __restrict__ s_sum,
                              const float* __restrict__ s_sq, const float* __restrict__ g,
                              const float* __restrict__ b, float* __restrict__ out) {
    int i = blockIdx.x * blockDim.x + threadIdx.x;
    if (i >= N_NODES * H_DIM) return;
    int c = i & 63;
    float mu  = s_sum[c] * (1.0f / N_NODES);
    float var = s_sq[c] * (1.0f / N_NODES) - mu * mu;
    float y = (in[i] - mu) * rsqrtf(var + BN_EPS) * g[c] + b[c];
    out[i] = fmaxf(y, 0.0f);
}

// ---------------- graph pooling (h >= 0 post-ReLU, so int-bit max & zero init are valid) ----------------
__global__ void k_pool(const float* __restrict__ h, const int* __restrict__ batch,
                       float* __restrict__ pcnt, float* __restrict__ psum, float* __restrict__ pmax) {
    int i = blockIdx.x * blockDim.x + threadIdx.x;
    if (i >= N_NODES * H_DIM) return;
    int n = i >> 6, c = i & 63;
    int g = batch[n];
    float v = h[i];
    atomicAdd(&psum[g * 64 + c], v);
    atomicMax((int*)&pmax[g * 64 + c], __float_as_int(v));
    if (c == 0) atomicAdd(&pcnt[g], 1.0f);
}

__global__ void k_head(const float* __restrict__ pcnt, const float* __restrict__ psum,
                       const float* __restrict__ pmax, const float* __restrict__ W,
                       const float* __restrict__ b, float* __restrict__ out) {
    int t = blockIdx.x * blockDim.x + threadIdx.x;
    if (t >= N_GRAPH * N_CLS) return;
    int g = t / N_CLS, k = t % N_CLS;
    float inv = 1.0f / fmaxf(pcnt[g], 1.0f);
    float acc = b[k];
#pragma unroll 4
    for (int j = 0; j < 64; ++j) {
        float sm = psum[g * 64 + j];
        acc += (sm * inv)      * W[(j)       * 3 + k];  // mean block
        acc += pmax[g * 64 + j]* W[(64 + j)  * 3 + k];  // max block
        acc += sm              * W[(128 + j) * 3 + k];  // sum block
    }
    out[t] = acc;
}

// ---------------- host side ----------------
static inline dim3 grid1(long n) { return dim3((unsigned)((n + 255) / 256)); }

extern "C" void kernel_launch(void* const* d_in, const int* in_sizes, int n_in,
                              void* d_out, int out_size, void* d_ws, size_t ws_size,
                              hipStream_t stream) {
    const float* x     = (const float*)d_in[0];
    const int*   ei    = (const int*)d_in[1];
    const int*   batch = (const int*)d_in[2];
    const float* W_exp = (const float*)d_in[3];
    const float* b_exp = (const float*)d_in[4];
    const float* gW1   = (const float*)d_in[5];
    const float* gb1   = (const float*)d_in[6];
    const float* gW2   = (const float*)d_in[7];
    const float* gb2   = (const float*)d_in[8];
    const float* W2    = (const float*)d_in[9];
    const float* b2    = (const float*)d_in[10];
    const float* bn1g  = (const float*)d_in[11];
    const float* bn1b  = (const float*)d_in[12];
    const float* bn2g  = (const float*)d_in[13];
    const float* bn2b  = (const float*)d_in[14];
    const float* clsW  = (const float*)d_in[15];
    const float* clsb  = (const float*)d_in[16];
    float* out = (float*)d_out;
    const int* esrc = ei;
    const int* edst = ei + N_EDGES;

    char* ws = (char*)d_ws;
    size_t off = 0;
    auto alloc = [&](size_t bytes) -> char* {
        char* p = ws + off;
        off += (bytes + 255) & ~(size_t)255;
        return p;
    };

    unsigned short* xb    = (unsigned short*)alloc((size_t)N_NODES * F_IN * 2);
    unsigned short* wexpt = (unsigned short*)alloc((size_t)N_EXP * H_DIM * F_IN * 2);
    unsigned short* gw1t  = (unsigned short*)alloc((size_t)32 * F_IN * 2);
    unsigned short* w2t   = (unsigned short*)alloc((size_t)H_DIM * H_DIM * 2);
    float* deg   = (float*)alloc((size_t)N_NODES * 4);
    float* dis   = (float*)alloc((size_t)N_NODES * 4);
    float* dinv  = (float*)alloc((size_t)N_NODES * 4);
    float* enorm = (float*)alloc((size_t)N_EDGES * 4);
    float* g1    = (float*)alloc((size_t)N_NODES * 32 * 4);   // reused as h2bf (same size)
    float* gate  = (float*)alloc((size_t)N_NODES * 3 * 4);
    float* h1    = (float*)alloc((size_t)N_NODES * 192 * 4);  // later sliced: h3 | agg2 | h5
    float* agg1  = (float*)alloc((size_t)N_NODES * 192 * 4);  // later reused as h4
    float* hc    = (float*)alloc((size_t)N_NODES * 64 * 4);
    float* stats = (float*)alloc(256 * 4);                    // sum1|sq1|sum2|sq2
    float* pcnt  = (float*)alloc(64 * 4);
    float* psum  = (float*)alloc((size_t)64 * 64 * 4);
    float* pmax  = (float*)alloc((size_t)64 * 64 * 4);

    unsigned short* h2bf = (unsigned short*)g1;
    float* h3   = h1;
    float* agg2 = h1 + (size_t)N_NODES * 64;
    float* h5   = h1 + (size_t)N_NODES * 128;
    float* h4   = agg1;

    const int MT = N_NODES / 16;                 // 3125 row-tiles (exact)
    const dim3 gGemm((unsigned)((MT + 7) / 8));  // 8 waves / 256-thread block

    // --- zero accumulators (ws is poisoned, not re-zeroed between replays) ---
    k_zero<<<grid1(N_NODES), 256, 0, stream>>>(deg, N_NODES);
    k_zero<<<grid1((long)N_NODES * 192), 256, 0, stream>>>(agg1, (long)N_NODES * 192);
    k_zero<<<grid1(256), 256, 0, stream>>>(stats, 256);
    k_zero<<<grid1(64 + 4096 + 4096), 256, 0, stream>>>(pcnt, 64 + 4096 + 4096); // pcnt|psum|pmax contiguous

    // --- precision / layout prep ---
    k_cvt_bf16<<<grid1((long)N_NODES * F_IN), 256, 0, stream>>>(x, xb, (long)N_NODES * F_IN);
    k_prep_wexp<<<grid1(N_EXP * F_IN * H_DIM), 256, 0, stream>>>(W_exp, wexpt);
    k_prep_gw1<<<grid1(F_IN * 32), 256, 0, stream>>>(gW1, gw1t);
    k_prep_w2<<<grid1(H_DIM * H_DIM), 256, 0, stream>>>(W2, w2t);

    // --- degrees & edge norms ---
    k_edge_deg<<<grid1(N_EDGES), 256, 0, stream>>>(edst, deg);
    k_deg_final<<<grid1(N_NODES), 256, 0, stream>>>(deg, dis, dinv);
    k_enorm<<<grid1(N_EDGES), 256, 0, stream>>>(esrc, edst, dis, enorm);

    // --- gate: GEMM1 (+bias+relu) then 32x3 matvec + softmax ---
    k_wmma_gemm<2, 4><<<gGemm, 256, 0, stream>>>(xb, F_IN, gw1t, F_IN, g1, 32, MT, gb1, 1);
    k_gate2<<<grid1(N_NODES), 256, 0, stream>>>(g1, gW2, gb2, gate);

    // --- expert GEMMs: h1[n][e*64+h] = x @ W_exp[e] ---
    for (int e = 0; e < N_EXP; ++e)
        k_wmma_gemm<4, 4><<<gGemm, 256, 0, stream>>>(
            xb, F_IN, wexpt + (size_t)e * H_DIM * F_IN, F_IN, h1 + e * 64, 192, MT, nullptr, 0);

    // --- GCN prop #1 (192 ch), gated combine, BN1+ReLU -> bf16 ---
    k_scatter<<<grid1((long)N_EDGES * 48), 256, 0, stream>>>(esrc, edst, enorm, h1, agg1, 192);
    k_combine1<<<grid1((long)N_NODES * 64), 256, 0, stream>>>(agg1, h1, gate, dinv, b_exp, hc);
    k_bnstats<<<dim3(192), 256, 0, stream>>>(hc, stats + 0, stats + 64);
    k_bnapply_bf16<<<grid1((long)N_NODES * 64), 256, 0, stream>>>(hc, stats + 0, stats + 64, bn1g, bn1b, h2bf);

    // --- layer 2: GEMM, prop #2 (64 ch), BN2+ReLU ---
    k_zero<<<grid1((long)N_NODES * 64), 256, 0, stream>>>(agg2, (long)N_NODES * 64);
    k_wmma_gemm<4, 2><<<gGemm, 256, 0, stream>>>(h2bf, H_DIM, w2t, H_DIM, h3, H_DIM, MT, nullptr, 0);
    k_scatter<<<grid1((long)N_EDGES * 16), 256, 0, stream>>>(esrc, edst, enorm, h3, agg2, 64);
    k_combine2<<<grid1((long)N_NODES * 64), 256, 0, stream>>>(agg2, h3, dinv, b2, h4);
    k_bnstats<<<dim3(192), 256, 0, stream>>>(h4, stats + 128, stats + 192);
    k_bnapply_f32<<<grid1((long)N_NODES * 64), 256, 0, stream>>>(h4, stats + 128, stats + 192, bn2g, bn2b, h5);

    // --- pooling + classifier head ---
    k_pool<<<grid1((long)N_NODES * 64), 256, 0, stream>>>(h5, batch, pcnt, psum, pmax);
    k_head<<<dim3(1), 256, 0, stream>>>(pcnt, psum, pmax, clsW, clsb, out);
}
